// ModuleCorrelation_9079560864093
// MI455X (gfx1250) — matrix operand order repaired and checked by reading, not compiled, not run on hardware
//
#include <hip/hip_runtime.h>
#include <hip/hip_bf16.h>

typedef _Float16 v8h  __attribute__((ext_vector_type(8)));
typedef _Float16 v16h __attribute__((ext_vector_type(16)));
typedef float    v8f  __attribute__((ext_vector_type(8)));

#define C_CH    128
#define H_IMG   160
#define W_IMG   224
#define B_N     8
#define NDISP   9          // 2*MAX_DISP+1
#define H_TILE  4
#define W_TILE  32
#define SROWS   (H_TILE + 8)   // 12 second rows (dy halo)
#define SCOLS   (W_TILE + 8)   // 40 second cols (dx halo)
#define CPAD    136            // 128 ch + 8 halves pad -> 272B stride, bank-conflict-free b128
#define SG_STRIDE 25           // 24 G columns + 1 pad

__device__ __forceinline__ v16h join16(v8h lo, v8h hi) {
  return __builtin_shufflevector(lo, hi, 0,1,2,3,4,5,6,7,8,9,10,11,12,13,14,15);
}

__global__ __launch_bounds__(256, 1)
void corr_wmma_f16_kernel(const float* __restrict__ x, float* __restrict__ out) {
  extern __shared__ char smem_raw[];
  _Float16* sSec = (_Float16*)smem_raw;                      // SROWS*SCOLS*CPAD halves
  _Float16* sFst = sSec + SROWS * SCOLS * CPAD;              // H_TILE*W_TILE*CPAD halves
  float*    sG   = (float*)(sFst + H_TILE * W_TILE * CPAD);  // 8 waves * 16 * SG_STRIDE floats

  const int b   = blockIdx.z;
  const int h0  = blockIdx.y * H_TILE;
  const int w0  = blockIdx.x * W_TILE;
  const int tid = threadIdx.x;

  const float* fst = x + ((size_t)b * 2 + 0) * C_CH * H_IMG * W_IMG;
  const float* sec = x + ((size_t)b * 2 + 1) * C_CH * H_IMG * W_IMG;

  // ---- Stage `second` (with zero halo), converting f32 -> f16 once for all 8 waves ----
  for (int idx = tid; idx < SROWS * C_CH * SCOLS; idx += 256) {
    int col = idx % SCOLS;
    int c   = (idx / SCOLS) % C_CH;
    int r   = idx / (SCOLS * C_CH);
    int gh  = h0 + r - 4;
    int gw  = w0 + col - 4;
    float v = 0.0f;
    if ((unsigned)gh < (unsigned)H_IMG && (unsigned)gw < (unsigned)W_IMG)
      v = sec[((size_t)c * H_IMG + gh) * W_IMG + gw];
    sSec[(r * SCOLS + col) * CPAD + c] = (_Float16)v;
  }
  // ---- Stage `first` ----
  for (int idx = tid; idx < H_TILE * C_CH * W_TILE; idx += 256) {
    int col = idx % W_TILE;
    int c   = (idx / W_TILE) % C_CH;
    int r   = idx / (W_TILE * C_CH);
    float v = fst[((size_t)c * H_IMG + (h0 + r)) * W_IMG + (w0 + col)];
    sFst[(r * W_TILE + col) * CPAD + c] = (_Float16)v;
  }
  __syncthreads();

  const int wave = tid >> 5;        // 0..7
  const int lane = tid & 31;
  const int nrow = lane & 15;       // M (A) or N (B/D) index
  const int hi16 = lane >> 4;       // lane half selects K sub-range
  const int hloc = wave >> 1;       // 0..3  output row within tile
  const int wsub = (wave & 1) * 16; // 0/16  output col strip within tile

  // ---- A fragments (16 pixels x K=128), reused across all 9 dy ----
  // ISA 16-bit A 16x32 layout: lanes0-15 hold K {0-7,16-23}, lanes16-31 hold K {8-15,24-31}
  v16h afrag[4];
#pragma unroll
  for (int kk = 0; kk < 4; ++kk) {
    int koff = kk * 32 + hi16 * 8;
    const _Float16* p = &sFst[(hloc * W_TILE + wsub + nrow) * CPAD + koff];
    afrag[kk] = join16(*(const v8h*)p, *(const v8h*)(p + 16));
  }

  float* sg = sG + wave * 16 * SG_STRIDE;

  for (int d = 0; d < NDISP; ++d) {
    const int r = hloc + d;          // second row in LDS (global h0+hloc+d-4)
    v8f g0 = {};
    v8f g1 = {};
#pragma unroll
    for (int kk = 0; kk < 4; ++kk) {
      int koffB = kk * 32 + hi16 * 8;  // B 32x16: lane column nrow, K {0-7,16-23} / {8-15,24-31}
      const _Float16* p0 = &sSec[(r * SCOLS + wsub + nrow) * CPAD + koffB];
      const _Float16* p1 = &sSec[(r * SCOLS + wsub + 8 + nrow) * CPAD + koffB];
      v16h b0 = join16(*(const v8h*)p0, *(const v8h*)(p0 + 16));
      v16h b1 = join16(*(const v8h*)p1, *(const v8h*)(p1 + 16));
      g0 = __builtin_amdgcn_wmma_f32_16x16x32_f16(false, afrag[kk], false, b0,
                                                  (short)0, g0, false, false);
      g1 = __builtin_amdgcn_wmma_f32_16x16x32_f16(false, afrag[kk], false, b1,
                                                  (short)0, g1, false, false);
    }
    // ---- Scatter D tiles to per-wave LDS scratch (C/D layout: VGPR rr -> M=rr(+8)) ----
#pragma unroll
    for (int rr = 0; rr < 8; ++rr) {
      int M = rr + hi16 * 8;
      sg[M * SG_STRIDE + nrow]     = g0[rr];   // G columns 0..15
      sg[M * SG_STRIDE + 8 + nrow] = g1[rr];   // G columns 8..23 (overlap identical)
    }
    // ---- Gather diagonal bands out[w,dx] = G[M, M+dx] / C, coalesced stores ----
    for (int t = lane; t < 144; t += 32) {
      int dx = t >> 4;
      int M  = t & 15;
      float v = sg[M * SG_STRIDE + M + dx] * (1.0f / 128.0f);
      size_t o = (((size_t)b * (NDISP * NDISP) + (d * NDISP + dx)) * H_IMG + (h0 + hloc))
                 * W_IMG + (w0 + wsub + M);
      out[o] = v;
    }
  }
}

extern "C" void kernel_launch(void* const* d_in, const int* in_sizes, int n_in,
                              void* d_out, int out_size, void* d_ws, size_t ws_size,
                              hipStream_t stream) {
  const float* x = (const float*)d_in[0];
  float* out = (float*)d_out;

  size_t shmem = (size_t)(SROWS * SCOLS * CPAD + H_TILE * W_TILE * CPAD) * sizeof(_Float16)
               + (size_t)8 * 16 * SG_STRIDE * sizeof(float);   // = 178,176 B (< 320 KB WGP LDS)
  hipFuncSetAttribute((const void*)corr_wmma_f16_kernel,
                      hipFuncAttributeMaxDynamicSharedMemorySize, (int)shmem);

  dim3 grid(W_IMG / W_TILE, H_IMG / H_TILE, B_N);  // 7 x 40 x 8
  dim3 block(256);                                  // 8 wave32
  corr_wmma_f16_kernel<<<grid, block, shmem, stream>>>(x, out);
}
